// NHGE_75273596830325
// MI455X (gfx1250) — compile-verified, matrix operand order checked by reading, not run on hardware
//
#include <hip/hip_runtime.h>
#include <math.h>

// ---------------- problem constants (match reference) ----------------
#define BATCH   2
#define NCTX    1024
#define DMODEL  256
#define NHEAD   8
#define FFDIM   1024
#define NLAYER  2
#define NITER   2
#define VOCAB   32000
#define DHEAD   32

typedef __attribute__((ext_vector_type(16))) __bf16 v16bf;
typedef __attribute__((ext_vector_type(8)))  float  v8f;
typedef __attribute__((ext_vector_type(4)))  int    v4i;

#if defined(__has_builtin)
#if __has_builtin(__builtin_amdgcn_global_load_async_to_lds_b128) && \
    __has_builtin(__builtin_amdgcn_s_wait_asynccnt)
#define HAVE_ASYNC_LDS 1
#endif
#endif

#ifdef HAVE_ASYNC_LDS
typedef __attribute__((address_space(1))) v4i gv4i;   // global int4
typedef __attribute__((address_space(3))) v4i lv4i;   // LDS int4
__device__ __forceinline__ void async_copy16(const void* g, void* l) {
    // low 32 bits of a generic LDS pointer are the LDS byte offset (ISA 10.2)
    gv4i* gp = (gv4i*)(unsigned long long)(size_t)g;
    lv4i* lp = (lv4i*)(unsigned)(size_t)l;
    __builtin_amdgcn_global_load_async_to_lds_b128(gp, lp, 0, 0);
}
#define WAIT_ASYNC() __builtin_amdgcn_s_wait_asynccnt(0)
#else
#define WAIT_ASYNC() ((void)0)
#endif

// ---------------- small helpers ----------------
__device__ __forceinline__ __bf16 f2bf(float f) {
    unsigned u = __builtin_bit_cast(unsigned, f);
    u += 0x7FFFu + ((u >> 16) & 1u);            // round-to-nearest-even
    unsigned short s = (unsigned short)(u >> 16);
    return __builtin_bit_cast(__bf16, s);
}

// Per CDNA5 ISA 7.12.2 the per-lane 16-bit A/B fragment is two contiguous
// 8-element K-chunks at [koff] and [koff+16], koff = (lane<16 ? 0 : 8).
// Load them as two b128s.
__device__ __forceinline__ v16bf ldfrag(const __bf16* p) {
    union { uint4 q[2]; v16bf v; } u;
    u.q[0] = *(const uint4*)p;
    u.q[1] = *(const uint4*)(p + 16);
    return u.v;
}

__device__ __forceinline__ float block_sum256(float v, float* sm) {
    int lane = threadIdx.x & 31, w = threadIdx.x >> 5;
    #pragma unroll
    for (int off = 16; off > 0; off >>= 1) v += __shfl_xor(v, off, 32);
    if (lane == 0) sm[w] = v;
    __syncthreads();
    float r = (threadIdx.x < 8) ? sm[threadIdx.x] : 0.0f;
    if (w == 0) {
        #pragma unroll
        for (int off = 4; off > 0; off >>= 1) r += __shfl_xor(r, off, 32);
        if (lane == 0) sm[0] = r;
    }
    __syncthreads();
    float out = sm[0];
    __syncthreads();
    return out;
}

// ---------------- kernels ----------------

// fp32 (K x N row-major) -> bf16 transposed (N x K): weights become B^T so both
// WMMA operands read K-contiguous fragments.
__global__ void to_bf16_T_kernel(const float* __restrict__ src, __bf16* __restrict__ dst,
                                 int K, int N) {
    int idx = blockIdx.x * blockDim.x + threadIdx.x;
    if (idx >= K * N) return;
    int k = idx / N, n = idx - k * N;
    dst[(size_t)n * K + k] = f2bf(src[idx]);
}

__global__ void embed_kernel(const int* __restrict__ ids, const float* __restrict__ tok,
                             const float* __restrict__ pos, float* __restrict__ hf) {
    int r = blockIdx.x, t = threadIdx.x;
    int n = r & (NCTX - 1);
    hf[r * DMODEL + t] = tok[(size_t)ids[r] * DMODEL + t] + pos[n * DMODEL + t];
}

__global__ void add_ln_kernel(float* __restrict__ hf, const float* __restrict__ delta,
                              const float* __restrict__ g, const float* __restrict__ bta,
                              __bf16* __restrict__ hb) {
    __shared__ float sm[8];
    int r = blockIdx.x, t = threadIdx.x;
    float x = hf[r * DMODEL + t];
    if (delta) x += delta[r * DMODEL + t];
    float mean = block_sum256(x, sm) * (1.0f / DMODEL);
    float d = x - mean;
    float var = block_sum256(d * d, sm) * (1.0f / DMODEL);
    float y = d * rsqrtf(var + 1e-5f) * g[t] + bta[t];
    hf[r * DMODEL + t] = y;
    if (hb) hb[r * DMODEL + t] = f2bf(y);
}

// C[M x Nc] = A[M x K]bf16 @ (Bt[Nc x K])^T + bias, outputs: f32 / bf16 / bf16-transposed.
// 128 threads = 4 waves; 64x64 block tile; wave owns 32x32 -> 4 WMMAs per 32-K step.
// Double-buffered LDS: async-DMA prefetch of tile k+32 overlaps WMMAs on tile k.
#define GP 40   // LDS pitch in bf16 elems (80B rows, 16B aligned chunks)
__global__ __launch_bounds__(128)
void gemm_bf16_kernel(const __bf16* __restrict__ A, const __bf16* __restrict__ Bt,
                      const float* __restrict__ bias, float* __restrict__ Cf,
                      __bf16* __restrict__ Cb, __bf16* __restrict__ CbT,
                      int M, int Nc, int K, int act) {
    __shared__ __bf16 As0[64 * GP], Bs0[64 * GP];
    __shared__ __bf16 As1[64 * GP], Bs1[64 * GP];
    const int t = threadIdx.x;
    const int lane = t & 31, w = t >> 5;
    const int wr = (w >> 1) * 32, wc = (w & 1) * 32;
    const int row0 = blockIdx.y * 64, col0 = blockIdx.x * 64;
    const int koff = (lane < 16) ? 0 : 8;
    const int m = lane & 15;

    v8f acc00 = {}, acc01 = {}, acc10 = {}, acc11 = {};

    // stage A (64x32) and Bt (64x32): 256 16B chunks each, 2 per thread
    auto stage = [&](__bf16* Asb, __bf16* Bsb, int k0s) {
        #pragma unroll
        for (int i = 0; i < 2; ++i) {
            int c = t + i * 128;
            int r = c >> 2, co = (c & 3) * 8;
#ifdef HAVE_ASYNC_LDS
            async_copy16(&A [(size_t)(row0 + r) * K + k0s + co], &Asb[r * GP + co]);
            async_copy16(&Bt[(size_t)(col0 + r) * K + k0s + co], &Bsb[r * GP + co]);
#else
            *(uint4*)&Asb[r * GP + co] = *(const uint4*)&A [(size_t)(row0 + r) * K + k0s + co];
            *(uint4*)&Bsb[r * GP + co] = *(const uint4*)&Bt[(size_t)(col0 + r) * K + k0s + co];
#endif
        }
    };
    auto compute = [&](const __bf16* Asb, const __bf16* Bsb) {
        v16bf a0 = ldfrag(&Asb[(wr      + m) * GP + koff]);
        v16bf a1 = ldfrag(&Asb[(wr + 16 + m) * GP + koff]);
        v16bf b0 = ldfrag(&Bsb[(wc      + m) * GP + koff]);
        v16bf b1 = ldfrag(&Bsb[(wc + 16 + m) * GP + koff]);
        acc00 = __builtin_amdgcn_wmma_f32_16x16x32_bf16(false, a0, false, b0, (short)0, acc00, false, false);
        acc01 = __builtin_amdgcn_wmma_f32_16x16x32_bf16(false, a0, false, b1, (short)0, acc01, false, false);
        acc10 = __builtin_amdgcn_wmma_f32_16x16x32_bf16(false, a1, false, b0, (short)0, acc10, false, false);
        acc11 = __builtin_amdgcn_wmma_f32_16x16x32_bf16(false, a1, false, b1, (short)0, acc11, false, false);
    };

    stage(As0, Bs0, 0);
    for (int k0 = 0; k0 < K; k0 += 64) {          // K % 64 == 0 here (256, 1024)
        WAIT_ASYNC();                              // tile k0 landed in buf0
        __syncthreads();                           // all waves done with buf1 reads
        stage(As1, Bs1, k0 + 32);                  // prefetch overlaps compute
        compute(As0, Bs0);

        WAIT_ASYNC();                              // tile k0+32 landed in buf1
        __syncthreads();                           // all waves done with buf0 reads
        if (k0 + 64 < K) stage(As0, Bs0, k0 + 64);
        compute(As1, Bs1);
    }

    auto epi = [&](v8f acc, int rb, int cb) {
        int nn = col0 + cb + (lane & 15);
        float bv = bias ? bias[nn] : 0.0f;
        #pragma unroll
        for (int v = 0; v < 8; ++v) {
            int mm = row0 + rb + v + ((lane >> 4) << 3);
            float x = acc[v] + bv;
            if (act == 1) x = 0.5f * x * (1.0f + erff(x * 0.70710678118654752f));
            if (Cf)  Cf [(size_t)mm * Nc + nn] = x;
            if (Cb)  Cb [(size_t)mm * Nc + nn] = f2bf(x);
            if (CbT) CbT[(size_t)nn * M + mm]  = f2bf(x);
        }
    };
    epi(acc00, wr,      wc);
    epi(acc01, wr,      wc + 16);
    epi(acc10, wr + 16, wc);
    epi(acc11, wr + 16, wc + 16);
}

__global__ void phase_proj_kernel(const float* __restrict__ hf, const float* __restrict__ ppw,
                                  const float* __restrict__ ppb, float* __restrict__ ph) {
    int idx = blockIdx.x * blockDim.x + threadIdx.x;
    if (idx >= BATCH * NCTX * NHEAD) return;
    int hh = idx & 7, row = idx >> 3;
    float s = ppb[hh];
    for (int k = 0; k < DMODEL; ++k) s += hf[(size_t)row * DMODEL + k] * ppw[k * NHEAD + hh];
    ph[idx] = s;
}

// fused attention: scores via WMMA + phase-cos modulation, softmax over 64KB LDS strip,
// P@V via WMMA against transposed V. 1 wave / (b, head, 16-row query tile).
__global__ __launch_bounds__(32)
void attention_kernel(const __bf16* __restrict__ Q, const __bf16* __restrict__ Km,
                      const __bf16* __restrict__ Vt, const float* __restrict__ ph,
                      __bf16* __restrict__ O) {
    __shared__ float S[16 * NCTX];                 // 64 KB score strip
    const int lane = threadIdx.x;
    const int bid = blockIdx.x;
    const int qt = bid & 63, hh = (bid >> 6) & 7, b = bid >> 9;
    const int i0 = qt * 16;
    const int koff = (lane < 16) ? 0 : 8;
    const int m = lane & 15;
    const float scale = 0.17677669529663687f;      // 1/sqrt(32)

    v16bf aq = ldfrag(&Q[(size_t)(b * NCTX + i0 + m) * DMODEL + hh * DHEAD + koff]);
    float pr[8];
    #pragma unroll
    for (int v = 0; v < 8; ++v)
        pr[v] = ph[(size_t)(b * NCTX + i0 + v + ((lane >> 4) << 3)) * NHEAD + hh];

    for (int jt = 0; jt < NCTX / 16; ++jt) {
        int j0 = jt * 16;
        v16bf bk = ldfrag(&Km[(size_t)(b * NCTX + j0 + m) * DMODEL + hh * DHEAD + koff]);
        v8f sc = {};
        sc = __builtin_amdgcn_wmma_f32_16x16x32_bf16(false, aq, false, bk, (short)0, sc, false, false);
        float phj = ph[(size_t)(b * NCTX + j0 + m) * NHEAD + hh];
        #pragma unroll
        for (int v = 0; v < 8; ++v) {
            int mm = v + ((lane >> 4) << 3);
            float s = sc[v] * scale;
            s = s * (1.0f + cosf(pr[v] - phj)) * 0.5f;
            S[mm * NCTX + j0 + m] = s;
        }
    }
    __syncthreads();

    for (int r = 0; r < 16; ++r) {
        float mx = -1e30f;
        for (int j = lane; j < NCTX; j += 32) mx = fmaxf(mx, S[r * NCTX + j]);
        #pragma unroll
        for (int off = 16; off > 0; off >>= 1) mx = fmaxf(mx, __shfl_xor(mx, off, 32));
        float sum = 0.0f;
        for (int j = lane; j < NCTX; j += 32) {
            float e = __expf(S[r * NCTX + j] - mx);
            S[r * NCTX + j] = e;
            sum += e;
        }
        #pragma unroll
        for (int off = 16; off > 0; off >>= 1) sum += __shfl_xor(sum, off, 32);
        float inv = 1.0f / sum;
        for (int j = lane; j < NCTX; j += 32) S[r * NCTX + j] *= inv;
    }
    __syncthreads();

    v8f o0 = {}, o1 = {};
    for (int kk = 0; kk < NCTX; kk += 32) {
        const float* pS = &S[m * NCTX + kk + koff];
        v16bf ap;
        #pragma unroll
        for (int e = 0; e < 8; ++e) { ap[e] = f2bf(pS[e]); ap[e + 8] = f2bf(pS[16 + e]); }
        v16bf b0 = ldfrag(&Vt[(size_t)(hh * DHEAD      + m) * (BATCH * NCTX) + b * NCTX + kk + koff]);
        v16bf b1 = ldfrag(&Vt[(size_t)(hh * DHEAD + 16 + m) * (BATCH * NCTX) + b * NCTX + kk + koff]);
        o0 = __builtin_amdgcn_wmma_f32_16x16x32_bf16(false, ap, false, b0, (short)0, o0, false, false);
        o1 = __builtin_amdgcn_wmma_f32_16x16x32_bf16(false, ap, false, b1, (short)0, o1, false, false);
    }
    #pragma unroll
    for (int v = 0; v < 8; ++v) {
        int mm = i0 + v + ((lane >> 4) << 3);
        size_t base = (size_t)(b * NCTX + mm) * DMODEL + hh * DHEAD;
        O[base + m]      = f2bf(o0[v]);
        O[base + 16 + m] = f2bf(o1[v]);
    }
}

__global__ void pool_kernel(const float* __restrict__ hf, float* __restrict__ pooled) {
    int idx = blockIdx.x * blockDim.x + threadIdx.x;       // B*D = 512
    if (idx >= BATCH * DMODEL) return;
    int b = idx >> 8, d = idx & 255;
    float s = 0.0f;
    for (int n = 0; n < NCTX; ++n) s += hf[((size_t)b * NCTX + n) * DMODEL + d];
    pooled[idx] = s * (1.0f / NCTX);
}

__global__ void head_kernel(const float* __restrict__ pooled, const float* __restrict__ hw,
                            const float* __restrict__ hb, float* __restrict__ out) {
    int idx = blockIdx.x * blockDim.x + threadIdx.x;       // B*V
    if (idx >= BATCH * VOCAB) return;
    int b = idx / VOCAB, vv = idx - b * VOCAB;
    float s = hb[vv];
    for (int k = 0; k < DMODEL; ++k) s += pooled[b * DMODEL + k] * hw[(size_t)k * VOCAB + vv];
    out[idx] = s;
}

// ---------------- host orchestration ----------------
static inline char* bump(char*& p, size_t bytes) {
    char* r = p;
    p += (bytes + 255) & ~(size_t)255;
    return r;
}

extern "C" void kernel_launch(void* const* d_in, const int* in_sizes, int n_in,
                              void* d_out, int out_size, void* d_ws, size_t ws_size,
                              hipStream_t stream) {
    (void)in_sizes; (void)n_in; (void)out_size; (void)ws_size;
    const int*   input_ids = (const int*)  d_in[0];
    const float* tok_emb   = (const float*)d_in[1];
    const float* pos_emb   = (const float*)d_in[2];
    const float* en_g      = (const float*)d_in[3];
    const float* en_b      = (const float*)d_in[4];
    // d_in[5], d_in[6] (pi_w, pi_b): dead state -> skipped
    const float* Wq        = (const float*)d_in[7];
    const float* Wk        = (const float*)d_in[8];
    const float* Wv        = (const float*)d_in[9];
    const float* Wo        = (const float*)d_in[10];
    const float* pp_w      = (const float*)d_in[11];
    const float* pp_b      = (const float*)d_in[12];
    const float* f1_w      = (const float*)d_in[13];
    const float* f1_b      = (const float*)d_in[14];
    const float* f2_w      = (const float*)d_in[15];
    const float* f2_b      = (const float*)d_in[16];
    const float* ln1_g     = (const float*)d_in[17];
    const float* ln1_b     = (const float*)d_in[18];
    const float* ln2_g     = (const float*)d_in[19];
    const float* ln2_b     = (const float*)d_in[20];
    // d_in[21], d_in[22] (pu_w, pu_b): dead state -> skipped
    const float* on_g      = (const float*)d_in[23];
    const float* on_b      = (const float*)d_in[24];
    const float* head_w    = (const float*)d_in[25];
    const float* head_b    = (const float*)d_in[26];
    float* out = (float*)d_out;

    const int ROWS = BATCH * NCTX;            // 2048
    char* p = (char*)d_ws;
    float*  hf     = (float*) bump(p, (size_t)ROWS * DMODEL * 4);
    __bf16* hb     = (__bf16*)bump(p, (size_t)ROWS * DMODEL * 2);
    __bf16* qb     = (__bf16*)bump(p, (size_t)ROWS * DMODEL * 2);
    __bf16* kb     = (__bf16*)bump(p, (size_t)ROWS * DMODEL * 2);
    __bf16* vtb    = (__bf16*)bump(p, (size_t)DMODEL * ROWS * 2);   // V^T [256 x 2048]
    __bf16* attnb  = (__bf16*)bump(p, (size_t)ROWS * DMODEL * 2);
    __bf16* ffb    = (__bf16*)bump(p, (size_t)ROWS * FFDIM * 2);
    float*  tmpf   = (float*) bump(p, (size_t)ROWS * DMODEL * 4);
    float*  ph     = (float*) bump(p, (size_t)ROWS * NHEAD * 4);
    float*  pooled = (float*) bump(p, (size_t)BATCH * DMODEL * 4);
    __bf16* wqb    = (__bf16*)bump(p, (size_t)NLAYER * DMODEL * DMODEL * 2);  // transposed
    __bf16* wkb    = (__bf16*)bump(p, (size_t)NLAYER * DMODEL * DMODEL * 2);
    __bf16* wvb    = (__bf16*)bump(p, (size_t)NLAYER * DMODEL * DMODEL * 2);
    __bf16* wob    = (__bf16*)bump(p, (size_t)NLAYER * DMODEL * DMODEL * 2);
    __bf16* f1b_   = (__bf16*)bump(p, (size_t)NLAYER * DMODEL * FFDIM * 2);
    __bf16* f2b_   = (__bf16*)bump(p, (size_t)NLAYER * FFDIM * DMODEL * 2);

    // --- weights -> bf16, pre-transposed to N-major ---
    const int nDD = DMODEL * DMODEL, nDF = DMODEL * FFDIM;
    for (int l = 0; l < NLAYER; ++l) {
        to_bf16_T_kernel<<<(nDD + 255) / 256, 256, 0, stream>>>(Wq + (size_t)l * nDD, wqb + (size_t)l * nDD, DMODEL, DMODEL);
        to_bf16_T_kernel<<<(nDD + 255) / 256, 256, 0, stream>>>(Wk + (size_t)l * nDD, wkb + (size_t)l * nDD, DMODEL, DMODEL);
        to_bf16_T_kernel<<<(nDD + 255) / 256, 256, 0, stream>>>(Wv + (size_t)l * nDD, wvb + (size_t)l * nDD, DMODEL, DMODEL);
        to_bf16_T_kernel<<<(nDD + 255) / 256, 256, 0, stream>>>(Wo + (size_t)l * nDD, wob + (size_t)l * nDD, DMODEL, DMODEL);
        to_bf16_T_kernel<<<(nDF + 255) / 256, 256, 0, stream>>>(f1_w + (size_t)l * nDF, f1b_ + (size_t)l * nDF, DMODEL, FFDIM);
        to_bf16_T_kernel<<<(nDF + 255) / 256, 256, 0, stream>>>(f2_w + (size_t)l * nDF, f2b_ + (size_t)l * nDF, FFDIM, DMODEL);
    }

    // --- embedding + input LN ---
    embed_kernel<<<ROWS, DMODEL, 0, stream>>>(input_ids, tok_emb, pos_emb, hf);
    add_ln_kernel<<<ROWS, DMODEL, 0, stream>>>(hf, nullptr, en_g, en_b, hb);

    dim3 gemmBlk(128);
    dim3 gDD(DMODEL / 64, ROWS / 64);     // (4, 32)
    dim3 gDF(FFDIM / 64, ROWS / 64);      // (16, 32)

    for (int t = 0; t < NITER; ++t) {
        for (int l = 0; l < NLAYER; ++l) {
            const __bf16* wq = wqb + (size_t)l * nDD;
            const __bf16* wk = wkb + (size_t)l * nDD;
            const __bf16* wv = wvb + (size_t)l * nDD;
            const __bf16* wo = wob + (size_t)l * nDD;
            const __bf16* w1 = f1b_ + (size_t)l * nDF;
            const __bf16* w2 = f2b_ + (size_t)l * nDF;

            gemm_bf16_kernel<<<gDD, gemmBlk, 0, stream>>>(hb, wq, nullptr, nullptr, qb, nullptr,
                                                          ROWS, DMODEL, DMODEL, 0);
            gemm_bf16_kernel<<<gDD, gemmBlk, 0, stream>>>(hb, wk, nullptr, nullptr, kb, nullptr,
                                                          ROWS, DMODEL, DMODEL, 0);
            gemm_bf16_kernel<<<gDD, gemmBlk, 0, stream>>>(hb, wv, nullptr, nullptr, nullptr, vtb,
                                                          ROWS, DMODEL, DMODEL, 0);
            phase_proj_kernel<<<(ROWS * NHEAD + 255) / 256, 256, 0, stream>>>(
                hf, pp_w + (size_t)l * DMODEL * NHEAD, pp_b + l * NHEAD, ph);
            attention_kernel<<<BATCH * NHEAD * (NCTX / 16), 32, 0, stream>>>(qb, kb, vtb, ph, attnb);
            gemm_bf16_kernel<<<gDD, gemmBlk, 0, stream>>>(attnb, wo, nullptr, tmpf, nullptr, nullptr,
                                                          ROWS, DMODEL, DMODEL, 0);
            add_ln_kernel<<<ROWS, DMODEL, 0, stream>>>(hf, tmpf, ln1_g + l * DMODEL,
                                                       ln1_b + l * DMODEL, hb);
            gemm_bf16_kernel<<<gDF, gemmBlk, 0, stream>>>(hb, w1, f1_b + l * FFDIM, nullptr, ffb, nullptr,
                                                          ROWS, FFDIM, DMODEL, 1 /*gelu*/);
            gemm_bf16_kernel<<<gDD, gemmBlk, 0, stream>>>(ffb, w2, f2_b + l * DMODEL, tmpf, nullptr, nullptr,
                                                          ROWS, DMODEL, FFDIM, 0);
            add_ln_kernel<<<ROWS, DMODEL, 0, stream>>>(hf, tmpf, ln2_g + l * DMODEL,
                                                       ln2_b + l * DMODEL, hb);
        }
    }

    // --- final LN, mean pool, vocab head ---
    add_ln_kernel<<<ROWS, DMODEL, 0, stream>>>(hf, nullptr, on_g, on_b, nullptr);
    pool_kernel<<<(BATCH * DMODEL + 255) / 256, 256, 0, stream>>>(hf, pooled);
    head_kernel<<<(BATCH * VOCAB + 255) / 256, 256, 0, stream>>>(pooled, head_w, head_b, out);
}
